// UniGCRModel_4922032521731
// MI455X (gfx1250) — matrix operand order repaired and checked by reading, not compiled, not run on hardware
//
#include <hip/hip_runtime.h>
#include <hip/hip_bf16.h>
#include <math.h>

#define B_SZ      1024
#define HIST_LEN  200
#define D_SZ      64
#define NUM_ITEMS 100000
#define NTOT      (NUM_ITEMS + 1)       // 100001
#define NUM_CAT   8
#define CAT_VOCAB 1000
#define NUM_NUM   4
#define PREFIX    (NUM_CAT + NUM_NUM)   // 12
#define LN_EPS    1e-5f
#define N_TILES   ((NTOT + 15) / 16)    // 6251
#define M_TILES   (B_SZ / 16)           // 64
#define FRAG_ELEMS 1024                 // per tile: 2 kk * 32 lanes * 16 elems

typedef __attribute__((ext_vector_type(16))) __bf16         v16bf;
typedef __attribute__((ext_vector_type(16))) unsigned short v16us;
typedef __attribute__((ext_vector_type(8)))  float          v8f;

// ---------- bf16 split helpers (round-to-nearest-even) ----------
__device__ __forceinline__ unsigned short f32_to_bf16_rne(float f) {
    unsigned u = __builtin_bit_cast(unsigned, f);
    unsigned r = (u + 0x7FFFu + ((u >> 16) & 1u)) >> 16;
    return (unsigned short)r;
}
__device__ __forceinline__ float bf16bits_to_f32(unsigned short h) {
    unsigned u = ((unsigned)h) << 16;
    return __builtin_bit_cast(float, u);
}
__device__ __forceinline__ void split_bits(float f, unsigned short &h, unsigned short &l) {
    h = f32_to_bf16_rne(f);
    l = f32_to_bf16_rne(f - bf16bits_to_f32(h));
}
__device__ __forceinline__ void split_bf16(float f, __bf16 &hi, __bf16 &lo) {
    unsigned short h, l;
    split_bits(f, h, l);
    hi = __builtin_bit_cast(__bf16, h);
    lo = __builtin_bit_cast(__bf16, l);
}

// ---------------------------------------------------------------
// Kernel 1: per-batch-row "u" vector (tiny).
// ---------------------------------------------------------------
__global__ void __launch_bounds__(64) unigcr_prep(
    const int*   __restrict__ history,   const int*   __restrict__ cat_feats,
    const float* __restrict__ num_feats, const float* __restrict__ item_emb,
    const float* __restrict__ cat_emb,   const float* __restrict__ num_w,
    const float* __restrict__ num_b,     const float* __restrict__ mlp_w1,
    const float* __restrict__ mlp_b1,    const float* __restrict__ mlp_w2,
    const float* __restrict__ mlp_b2,    const float* __restrict__ ln_g,
    const float* __restrict__ ln_b,      const float* __restrict__ bb_w,
    const float* __restrict__ bb_b,      float* __restrict__ u_out)
{
    __shared__ float sh[64];
    __shared__ float sh2[128];
    __shared__ float vsh[64];
    __shared__ int   icnt[64];
    __shared__ int   cnt_sh;

    const int b = blockIdx.x;
    const int t = threadIdx.x;

    int c = 0;
    for (int j = t; j < HIST_LEN; j += 64)
        c += (history[b * HIST_LEN + j] != 0) ? 1 : 0;
    icnt[t] = c;
    __syncthreads();
    if (t == 0) {
        int s = 0;
        for (int i = 0; i < 64; ++i) s += icnt[i];
        cnt_sh = s;
    }
    __syncthreads();

    const int hist_len = cnt_sh;
    const int end_idx  = PREFIX + hist_len - 1;

    float v;
    if (end_idx >= PREFIX) {
        int item = history[b * HIST_LEN + (hist_len - 1)];
        v = item_emb[(size_t)item * D_SZ + t];
    } else {
        // Rare fallback: prefix token -> MLP + LayerNorm (exact semantics).
        int r = end_idx < 0 ? 0 : end_idx;
        float tok;
        if (r < NUM_CAT) {
            int cf = cat_feats[b * NUM_CAT + r];
            tok = cat_emb[((size_t)r * CAT_VOCAB + cf) * D_SZ + t];
        } else {
            int q = r - NUM_CAT;
            tok = num_feats[b * NUM_NUM + q] * num_w[q * D_SZ + t] + num_b[q * D_SZ + t];
        }
        sh[t] = tok;
        __syncthreads();
        float h1a = mlp_b1[t], h1b = mlp_b1[t + 64];
        for (int k = 0; k < 64; ++k) {
            float x = sh[k];
            h1a += x * mlp_w1[k * 128 + t];
            h1b += x * mlp_w1[k * 128 + t + 64];
        }
        h1a = 0.5f * h1a * (1.0f + erff(h1a * 0.70710678118f));
        h1b = 0.5f * h1b * (1.0f + erff(h1b * 0.70710678118f));
        __syncthreads();
        sh2[t] = h1a; sh2[t + 64] = h1b;
        __syncthreads();
        float h = mlp_b2[t];
        for (int j = 0; j < 128; ++j) h += sh2[j] * mlp_w2[j * 64 + t];
        __syncthreads();
        sh[t] = h;
        __syncthreads();
        float mu = 0.0f, sq = 0.0f;
        for (int i = 0; i < 64; ++i) { mu += sh[i]; sq += sh[i] * sh[i]; }
        mu *= (1.0f / 64.0f); sq *= (1.0f / 64.0f);
        float var = sq - mu * mu;
        v = (h - mu) * rsqrtf(var + LN_EPS) * ln_g[t] + ln_b[t];
    }

    vsh[t] = v;
    __syncthreads();
    float acc = bb_b[t];
    for (int k = 0; k < 64; ++k) acc += vsh[k] * bb_w[k * D_SZ + t];
    float uu = acc / (1.0f + expf(-acc));
    u_out[b * D_SZ + t] = uu;
}

// ---------------------------------------------------------------
// Kernel 2a: one-time split of u into WMMA A-fragment layout.
// A 16x32 bf16 layout: lane(0-15)=row M, kh=(lane>>4)*8;
//   elem e<8 -> K=kh+e, e>=8 -> K=16+kh+(e-8)   (+ kk*32)
// Storage: ((m_tile*2+kk)*32 + lane)*16 + e  (per-lane contiguous 32B)
// ---------------------------------------------------------------
__global__ void __launch_bounds__(32) unigcr_pack_a(
    const float* __restrict__ u,
    unsigned short* __restrict__ ahi, unsigned short* __restrict__ alo)
{
    const int m_tile = blockIdx.x;
    const int lane   = threadIdx.x;
    const int row    = m_tile * 16 + (lane & 15);
    const int khalf  = (lane >> 4) * 8;
#pragma unroll
    for (int kk = 0; kk < 2; ++kk) {
        v16us h, l;
#pragma unroll
        for (int e = 0; e < 8; ++e) {
            unsigned short hh, ll;
            split_bits(u[row * D_SZ + kk * 32 + khalf + e], hh, ll);
            h[e] = hh; l[e] = ll;
            split_bits(u[row * D_SZ + kk * 32 + 16 + khalf + e], hh, ll);
            h[8 + e] = hh; l[8 + e] = ll;
        }
        size_t idx = ((size_t)(m_tile * 2 + kk) * 32 + lane) * 16;
        *(v16us*)(ahi + idx) = h;
        *(v16us*)(alo + idx) = l;
    }
}

// ---------------------------------------------------------------
// Kernel 2b: one-time split of gr_w into WMMA B-fragment layout.
// B 32x16 bf16 layout: lane(0-15)=col N, kb=(lane>>4)*8; e<8 -> K=kb+e, e>=8 -> K=16+kb+(e-8) (+ kk*32).
// Tail columns (>= NTOT) zero-padded so GEMM loads are unconditional.
// ---------------------------------------------------------------
__global__ void __launch_bounds__(256) unigcr_pack_b(
    const float* __restrict__ gr_w,
    unsigned short* __restrict__ bhi, unsigned short* __restrict__ blo)
{
    const int lane   = threadIdx.x & 31;
    const int wave   = threadIdx.x >> 5;
    const int n_tile = blockIdx.x * 8 + wave;
    if (n_tile >= N_TILES) return;

    const int  ncol   = n_tile * 16 + (lane & 15);
    const bool nvalid = (ncol < NTOT);
    const int  kb     = (lane >> 4) * 8;
#pragma unroll
    for (int kk = 0; kk < 2; ++kk) {
        v16us h, l;
#pragma unroll
        for (int e = 0; e < 16; ++e) {
            int k = kk * 32 + kb + e + (e >= 8 ? 8 : 0);
            float f = nvalid ? gr_w[(size_t)k * NTOT + ncol] : 0.0f;
            unsigned short hh, ll;
            split_bits(f, hh, ll);
            h[e] = hh; l[e] = ll;
        }
        size_t idx = ((size_t)(n_tile * 2 + kk) * 32 + lane) * 16;
        *(v16us*)(bhi + idx) = h;
        *(v16us*)(blo + idx) = l;
    }
}

// ---------------------------------------------------------------
// Kernel 3 (fast): gr_logits = u @ gr_w + gr_b using pre-split
// fragments. Hot loop: 16 vector loads, 6 WMMA, 8 stores. No splits.
// ---------------------------------------------------------------
__global__ void __launch_bounds__(256) unigcr_grgemm_fast(
    const unsigned short* __restrict__ ahi_g, const unsigned short* __restrict__ alo_g,
    const unsigned short* __restrict__ bhi_g, const unsigned short* __restrict__ blo_g,
    const float* __restrict__ gr_b, float* __restrict__ logits)
{
    const int lane   = threadIdx.x & 31;
    const int wave   = threadIdx.x >> 5;
    const int m_tile = blockIdx.x;
    const int n_tile = blockIdx.y * 8 + wave;
    if (n_tile >= N_TILES) return;          // wave-uniform: EXEC stays all-1s

    v16bf ahi[2], alo[2], bhi[2], blo[2];
#pragma unroll
    for (int kk = 0; kk < 2; ++kk) {
        size_t aidx = ((size_t)(m_tile * 2 + kk) * 32 + lane) * 16;
        ahi[kk] = __builtin_bit_cast(v16bf, *(const v16us*)(ahi_g + aidx));
        alo[kk] = __builtin_bit_cast(v16bf, *(const v16us*)(alo_g + aidx));
        size_t bidx = ((size_t)(n_tile * 2 + kk) * 32 + lane) * 16;
        bhi[kk] = __builtin_bit_cast(v16bf, *(const v16us*)(bhi_g + bidx));
        blo[kk] = __builtin_bit_cast(v16bf, *(const v16us*)(blo_g + bidx));
    }

    v8f acc = {};
    acc = __builtin_amdgcn_wmma_f32_16x16x32_bf16(false, alo[0], false, bhi[0], (short)0, acc, false, false);
    acc = __builtin_amdgcn_wmma_f32_16x16x32_bf16(false, alo[1], false, bhi[1], (short)0, acc, false, false);
    acc = __builtin_amdgcn_wmma_f32_16x16x32_bf16(false, ahi[0], false, blo[0], (short)0, acc, false, false);
    acc = __builtin_amdgcn_wmma_f32_16x16x32_bf16(false, ahi[1], false, blo[1], (short)0, acc, false, false);
    acc = __builtin_amdgcn_wmma_f32_16x16x32_bf16(false, ahi[0], false, bhi[0], (short)0, acc, false, false);
    acc = __builtin_amdgcn_wmma_f32_16x16x32_bf16(false, ahi[1], false, bhi[1], (short)0, acc, false, false);

    const int ncol = n_tile * 16 + (lane & 15);
    if (ncol < NTOT) {
        float bias = gr_b[ncol];
        const int mh = (lane >> 4) * 8;
#pragma unroll
        for (int i = 0; i < 8; ++i) {
            int m = m_tile * 16 + mh + i;
            logits[(size_t)m * NTOT + ncol] = acc[i] + bias;
        }
    }
}

// ---------------------------------------------------------------
// Kernel 3 (fallback, ws too small): in-register split GEMM.
// ---------------------------------------------------------------
__global__ void __launch_bounds__(256) unigcr_grgemm_split(
    const float* __restrict__ u, const float* __restrict__ gr_w,
    const float* __restrict__ gr_b, float* __restrict__ logits)
{
    const int lane   = threadIdx.x & 31;
    const int wave   = threadIdx.x >> 5;
    const int m_tile = blockIdx.x;
    const int n_tile = blockIdx.y * 8 + wave;
    if (n_tile >= N_TILES) return;

    const int row   = m_tile * 16 + (lane & 15);
    const int khalf = (lane >> 4) * 8;

    v16bf ahi[2], alo[2], bhi[2], blo[2];
#pragma unroll
    for (int kk = 0; kk < 2; ++kk) {
        const float* a0 = u + row * D_SZ + kk * 32 + khalf;
        const float* a1 = a0 + 16;
#pragma unroll
        for (int e = 0; e < 8; ++e) {
            __bf16 hi, lo;
            split_bf16(a0[e], hi, lo);
            ahi[kk][e] = hi; alo[kk][e] = lo;
            split_bf16(a1[e], hi, lo);
            ahi[kk][8 + e] = hi; alo[kk][8 + e] = lo;
        }
    }
    const int  ncol   = n_tile * 16 + (lane & 15);
    const bool nvalid = (ncol < NTOT);
    const int  kb     = (lane >> 4) * 8;
#pragma unroll
    for (int kk = 0; kk < 2; ++kk) {
#pragma unroll
        for (int e = 0; e < 16; ++e) {
            int k  = kk * 32 + kb + e + (e >= 8 ? 8 : 0);
            float f = nvalid ? gr_w[(size_t)k * NTOT + ncol] : 0.0f;
            __bf16 hi, lo;
            split_bf16(f, hi, lo);
            bhi[kk][e] = hi; blo[kk][e] = lo;
        }
    }
    v8f acc = {};
    acc = __builtin_amdgcn_wmma_f32_16x16x32_bf16(false, alo[0], false, bhi[0], (short)0, acc, false, false);
    acc = __builtin_amdgcn_wmma_f32_16x16x32_bf16(false, alo[1], false, bhi[1], (short)0, acc, false, false);
    acc = __builtin_amdgcn_wmma_f32_16x16x32_bf16(false, ahi[0], false, blo[0], (short)0, acc, false, false);
    acc = __builtin_amdgcn_wmma_f32_16x16x32_bf16(false, ahi[1], false, blo[1], (short)0, acc, false, false);
    acc = __builtin_amdgcn_wmma_f32_16x16x32_bf16(false, ahi[0], false, bhi[0], (short)0, acc, false, false);
    acc = __builtin_amdgcn_wmma_f32_16x16x32_bf16(false, ahi[1], false, bhi[1], (short)0, acc, false, false);

    if (nvalid) {
        float bias = gr_b[ncol];
        const int mh = (lane >> 4) * 8;
#pragma unroll
        for (int i = 0; i < 8; ++i) {
            int m = m_tile * 16 + mh + i;
            logits[(size_t)m * NTOT + ncol] = acc[i] + bias;
        }
    }
}

// ---------------------------------------------------------------
extern "C" void kernel_launch(void* const* d_in, const int* in_sizes, int n_in,
                              void* d_out, int out_size, void* d_ws, size_t ws_size,
                              hipStream_t stream) {
    const int*   history   = (const int*)  d_in[0];
    const int*   cat_feats = (const int*)  d_in[1];
    const float* num_feats = (const float*)d_in[2];
    const float* item_emb  = (const float*)d_in[3];
    const float* cat_emb   = (const float*)d_in[4];
    const float* num_w     = (const float*)d_in[5];
    const float* num_b     = (const float*)d_in[6];
    const float* mlp_w1    = (const float*)d_in[7];
    const float* mlp_b1    = (const float*)d_in[8];
    const float* mlp_w2    = (const float*)d_in[9];
    const float* mlp_b2    = (const float*)d_in[10];
    const float* ln_g      = (const float*)d_in[11];
    const float* ln_b      = (const float*)d_in[12];
    const float* bb_w      = (const float*)d_in[13];
    const float* bb_b      = (const float*)d_in[14];
    const float* gr_w      = (const float*)d_in[15];
    const float* gr_b      = (const float*)d_in[16];

    float* out    = (float*)d_out;
    float* u_out  = out;                       // (1024, 64)
    float* logits = out + (size_t)B_SZ * D_SZ; // (1024, 100001)

    unigcr_prep<<<dim3(B_SZ), dim3(64), 0, stream>>>(
        history, cat_feats, num_feats, item_emb, cat_emb, num_w, num_b,
        mlp_w1, mlp_b1, mlp_w2, mlp_b2, ln_g, ln_b, bb_w, bb_b, u_out);

    const size_t bElems = (size_t)N_TILES * FRAG_ELEMS;   // 6,401,024
    const size_t aElems = (size_t)M_TILES * FRAG_ELEMS;   // 65,536
    const size_t need   = (2 * bElems + 2 * aElems) * sizeof(unsigned short);

    if (ws_size >= need) {
        unsigned short* bhi = (unsigned short*)d_ws;
        unsigned short* blo = bhi + bElems;
        unsigned short* ahi = blo + bElems;
        unsigned short* alo = ahi + aElems;

        unigcr_pack_b<<<dim3((N_TILES + 7) / 8), dim3(256), 0, stream>>>(gr_w, bhi, blo);
        unigcr_pack_a<<<dim3(M_TILES), dim3(32), 0, stream>>>(u_out, ahi, alo);
        unigcr_grgemm_fast<<<dim3(M_TILES, (N_TILES + 7) / 8), dim3(256), 0, stream>>>(
            ahi, alo, bhi, blo, gr_b, logits);
    } else {
        unigcr_grgemm_split<<<dim3(M_TILES, (N_TILES + 7) / 8), dim3(256), 0, stream>>>(
            u_out, gr_w, gr_b, logits);
    }
}
